// LongViT_8950711844882
// MI455X (gfx1250) — compile-verified, matrix operand order checked by reading, not run on hardware
//
#include <hip/hip_runtime.h>
#include <hip/hip_bf16.h>
#include <math.h>

#define IMGSZ 512
#define PSZ 16
#define CIN 3
#define DIM 512
#define DEPTHL 2
#define HEADS 8
#define HD 64
#define MLPD 2048
#define WBAND 64
#define NCLS 1000
#define BBATCH 8
#define NPATCH 1024
#define NTOK0 1025
#define NTOK 1152              // padded to multiple of 2*W
#define NCH (NTOK / WBAND)     // 18 chunks
#define KPATCH (PSZ * PSZ * CIN) // 768

typedef __attribute__((ext_vector_type(16))) __bf16 v16bf;
typedef __attribute__((ext_vector_type(8)))  float  v8f;

// ---------------------------------------------------------------------------
// Generic bf16 WMMA GEMM: C[M,N] (+)= A[M,K] @ B[K,N] + bias[N]  (optional GELU)
// Block = 256 threads = 8 waves arranged 4(row) x 2(col).
// Block tile 128x64, wave tile 32x32 (4 WMMAs / K-step), K-step 32,
// double-buffered LDS staging (one barrier per K-step).
// ---------------------------------------------------------------------------
template<int GELU_ACT, int ACCUM>
__global__ __launch_bounds__(256)
void gemm_bf16_wmma(const float* __restrict__ A, const float* __restrict__ Bw,
                    const float* __restrict__ bias, float* __restrict__ C,
                    int M, int K, int Nn)
{
    __shared__ __bf16 As[2][128][40];   // [buf][m][k], padded
    __shared__ __bf16 Bs[2][64][40];    // [buf][n][k] (transposed), padded

    const int tid  = threadIdx.x;
    const int lane = tid & 31;
    const int wid  = tid >> 5;
    const int mr   = wid & 3;       // wave row-pair tile: rows mr*32 .. mr*32+31
    const int ng   = wid >> 2;      // wave col-group 0..1 (32 cols each)
    const int lm   = lane & 15;
    const int kh   = lane >> 4;
    const int rowBase = blockIdx.y * 128;
    const int colBase = blockIdx.x * 64;

    auto stage = [&](int buf, int k0) {
        // A tile 128x32 (fp32 -> bf16), 16 elems/thread
        for (int i = tid; i < 128 * 32; i += 256) {
            int m = i >> 5, kk = i & 31;
            int gr = rowBase + m;
            float va = (gr < M) ? A[(size_t)gr * K + k0 + kk] : 0.f;
            As[buf][m][kk] = (__bf16)va;
        }
        // B tile 32x64, transposed into Bs[n][k], 8 elems/thread
        for (int i = tid; i < 32 * 64; i += 256) {
            int kk = i >> 6, n = i & 63;
            float vb = Bw[(size_t)(k0 + kk) * Nn + colBase + n];
            Bs[buf][n][kk] = (__bf16)vb;
        }
    };

    v8f acc00 = {}, acc01 = {}, acc10 = {}, acc11 = {};

    stage(0, 0);
    __syncthreads();

    const int nk = K >> 5;
    for (int kt = 0; kt < nk; ++kt) {
        const int buf = kt & 1;
        if (kt + 1 < nk) stage(buf ^ 1, (kt + 1) << 5);   // overlap next-tile staging

        v16bf a0, a1, b0, b1;
        {
            const __bf16* p0 = &As[buf][mr * 32 + lm][kh * 8];
            const __bf16* p1 = &As[buf][mr * 32 + lm][16 + kh * 8];
            #pragma unroll
            for (int e = 0; e < 8; ++e) { a0[e] = p0[e]; a0[8 + e] = p1[e]; }
            p0 = &As[buf][mr * 32 + 16 + lm][kh * 8];
            p1 = &As[buf][mr * 32 + 16 + lm][16 + kh * 8];
            #pragma unroll
            for (int e = 0; e < 8; ++e) { a1[e] = p0[e]; a1[8 + e] = p1[e]; }
        }
        {
            const __bf16* p0 = &Bs[buf][ng * 32 + lm][kh * 8];
            const __bf16* p1 = &Bs[buf][ng * 32 + lm][16 + kh * 8];
            #pragma unroll
            for (int e = 0; e < 8; ++e) { b0[e] = p0[e]; b0[8 + e] = p1[e]; }
            p0 = &Bs[buf][ng * 32 + 16 + lm][kh * 8];
            p1 = &Bs[buf][ng * 32 + 16 + lm][16 + kh * 8];
            #pragma unroll
            for (int e = 0; e < 8; ++e) { b1[e] = p0[e]; b1[8 + e] = p1[e]; }
        }
        acc00 = __builtin_amdgcn_wmma_f32_16x16x32_bf16(false, a0, false, b0, (short)0, acc00, false, false);
        acc01 = __builtin_amdgcn_wmma_f32_16x16x32_bf16(false, a0, false, b1, (short)0, acc01, false, false);
        acc10 = __builtin_amdgcn_wmma_f32_16x16x32_bf16(false, a1, false, b0, (short)0, acc10, false, false);
        acc11 = __builtin_amdgcn_wmma_f32_16x16x32_bf16(false, a1, false, b1, (short)0, acc11, false, false);
        __syncthreads();
    }

    // epilogue: 4 tiles of 16x16 per wave
    #pragma unroll
    for (int rt = 0; rt < 2; ++rt) {
        const v8f* ar0 = rt ? &acc10 : &acc00;
        const v8f* ar1 = rt ? &acc11 : &acc01;
        #pragma unroll
        for (int r = 0; r < 8; ++r) {
            int gm = rowBase + mr * 32 + rt * 16 + r + 8 * kh;
            if (gm >= M) continue;
            int gn0 = colBase + ng * 32 + lm;
            int gn1 = gn0 + 16;
            float v0 = (*ar0)[r] + bias[gn0];
            float v1 = (*ar1)[r] + bias[gn1];
            if (GELU_ACT) {
                v0 = 0.5f * v0 * (1.f + erff(v0 * 0.70710678118f));
                v1 = 0.5f * v1 * (1.f + erff(v1 * 0.70710678118f));
            }
            if (ACCUM) {
                C[(size_t)gm * Nn + gn0] += v0;
                C[(size_t)gm * Nn + gn1] += v1;
            } else {
                C[(size_t)gm * Nn + gn0] = v0;
                C[(size_t)gm * Nn + gn1] = v1;
            }
        }
    }
}

// ---------------------------------------------------------------------------
// im2col: img[B,C,512,512] -> xcol[B*1024, 768] per reference transpose order
// ---------------------------------------------------------------------------
__global__ void im2col_kernel(const float* __restrict__ img, float* __restrict__ xcol)
{
    size_t idx = (size_t)blockIdx.x * blockDim.x + threadIdx.x;
    size_t total = (size_t)BBATCH * NPATCH * KPATCH;
    if (idx >= total) return;
    int col = (int)(idx % KPATCH);
    size_t t = idx / KPATCH;
    int p = (int)(t % NPATCH);
    int b = (int)(t / NPATCH);
    int ch = col % CIN;
    int pc = (col / CIN) % PSZ;
    int pr = (col / CIN) / PSZ;
    int py = p >> 5, px = p & 31;
    int y = py * PSZ + pr, x = px * PSZ + pc;
    xcol[idx] = img[(((size_t)b * CIN + ch) * IMGSZ + y) * IMGSZ + x];
}

// assemble x[b,n,d] = cls/patch + pos_emb, zero padding rows
__global__ void assemble_kernel(const float* __restrict__ xpe, const float* __restrict__ cls_tok,
                                const float* __restrict__ pos_emb, float* __restrict__ x)
{
    size_t idx = (size_t)blockIdx.x * blockDim.x + threadIdx.x;
    size_t total = (size_t)BBATCH * NTOK * DIM;
    if (idx >= total) return;
    int d = (int)(idx % DIM);
    size_t t = idx / DIM;
    int n = (int)(t % NTOK);
    int b = (int)(t / NTOK);
    float v;
    if (n == 0)             v = cls_tok[d] + pos_emb[d];
    else if (n < NTOK0)     v = xpe[((size_t)b * NPATCH + n - 1) * DIM + d] + pos_emb[(size_t)n * DIM + d];
    else                    v = 0.f;
    x[idx] = v;
}

__global__ void add_kernel(float* __restrict__ x, const float* __restrict__ y, size_t total)
{
    size_t idx = (size_t)blockIdx.x * blockDim.x + threadIdx.x;
    if (idx < total) x[idx] += y[idx];
}

// ---------------------------------------------------------------------------
// LayerNorm over DIM=512, one block per row (256 threads, 2 elems each)
// ---------------------------------------------------------------------------
__global__ __launch_bounds__(256)
void layernorm_kernel(const float* __restrict__ x, const float* __restrict__ w,
                      const float* __restrict__ b, float* __restrict__ y)
{
    __shared__ float red[256];
    const int tid = threadIdx.x;
    const float* xr = x + (size_t)blockIdx.x * DIM;
    float v0 = xr[tid], v1 = xr[tid + 256];
    red[tid] = v0 + v1; __syncthreads();
    for (int o = 128; o > 0; o >>= 1) { if (tid < o) red[tid] += red[tid + o]; __syncthreads(); }
    float mean = red[0] * (1.f / DIM); __syncthreads();
    float d0 = v0 - mean, d1 = v1 - mean;
    red[tid] = d0 * d0 + d1 * d1; __syncthreads();
    for (int o = 128; o > 0; o >>= 1) { if (tid < o) red[tid] += red[tid + o]; __syncthreads(); }
    float rs = rsqrtf(red[0] * (1.f / DIM) + 1e-5f);
    float* yr = (float*)y + (size_t)blockIdx.x * DIM;
    yr[tid]       = d0 * rs * w[tid]       + b[tid];
    yr[tid + 256] = d1 * rs * w[tid + 256] + b[tid + 256];
}

// qg row-0 projection: qg[b,:] = xln[b,0,:] @ wqg + bqg   (8 x 512, tiny)
__global__ __launch_bounds__(256)
void qg_proj_kernel(const float* __restrict__ xln, const float* __restrict__ wqg,
                    const float* __restrict__ bqg, float* __restrict__ qg)
{
    const int b = blockIdx.x;
    __shared__ float xr[DIM];
    for (int i = threadIdx.x; i < DIM; i += 256) xr[i] = xln[(size_t)b * NTOK * DIM + i];
    __syncthreads();
    for (int j = threadIdx.x; j < DIM; j += 256) {
        float acc = bqg[j];
        for (int kk = 0; kk < DIM; ++kk) acc += xr[kk] * wqg[(size_t)kk * DIM + j];
        qg[b * DIM + j] = acc;
    }
}

// ---------------------------------------------------------------------------
// Banded local attention: one block per (chunk c, head h, batch b).
// S = (q*sc) @ k3^T via WMMA; masked softmax over [gcol | 192 band]; out via WMMA.
// ---------------------------------------------------------------------------
__global__ __launch_bounds__(256)
void band_attn_kernel(const float* __restrict__ q, const float* __restrict__ k,
                      const float* __restrict__ v, float* __restrict__ out)
{
    const int c = blockIdx.x, h = blockIdx.y, b = blockIdx.z;
    __shared__ __bf16 qs[64][72];
    __shared__ __bf16 ks[192][72];
    __shared__ __bf16 vs[192][72];
    __shared__ float  S[64][200];   // col 0 = gcol, cols 1..192 = band
    __shared__ float  k0s[64];
    __shared__ float  v0s[64];

    const int tid = threadIdx.x, lane = tid & 31, wid = tid >> 5;
    const int lm = lane & 15, kh = lane >> 4;
    const size_t base = (size_t)b * NTOK * DIM + (size_t)h * HD;

    for (int i = tid; i < 64 * 64; i += 256) {
        int r = i >> 6, d = i & 63;
        qs[r][d] = (__bf16)(q[base + (size_t)(c * WBAND + r) * DIM + d] * 0.125f);
    }
    for (int i = tid; i < 192 * 64; i += 256) {
        int j = i >> 6, d = i & 63;
        int cc = c - 1 + (j >> 6);
        int rr = j & 63;
        float kv = 0.f, vv = 0.f;
        if (cc >= 0 && cc < NCH) {
            size_t off = base + (size_t)(cc * WBAND + rr) * DIM + d;
            kv = k[off]; vv = v[off];
        }
        ks[j][d] = (__bf16)kv;
        vs[j][d] = (__bf16)vv;
    }
    if (tid < 64) { k0s[tid] = k[base + tid]; v0s[tid] = v[base + tid]; }
    __syncthreads();

    // ---- scores: 64x192 = 4x12 tiles; wave (mr,cg) does 6 col tiles ----
    {
        const int mr = wid & 3;
        const int cg = wid >> 2;
        v16bf af[2];
        #pragma unroll
        for (int s2 = 0; s2 < 2; ++s2) {
            const __bf16* p0 = &qs[mr * 16 + lm][s2 * 32 + kh * 8];
            const __bf16* p1 = &qs[mr * 16 + lm][s2 * 32 + 16 + kh * 8];
            #pragma unroll
            for (int e = 0; e < 8; ++e) { af[s2][e] = p0[e]; af[s2][8 + e] = p1[e]; }
        }
        for (int ct = 0; ct < 6; ++ct) {
            int jt = (cg * 6 + ct) * 16;
            v8f acc = {};
            #pragma unroll
            for (int s2 = 0; s2 < 2; ++s2) {
                v16bf bfr;
                const __bf16* p0 = &ks[jt + lm][s2 * 32 + kh * 8];
                const __bf16* p1 = &ks[jt + lm][s2 * 32 + 16 + kh * 8];
                #pragma unroll
                for (int e = 0; e < 8; ++e) { bfr[e] = p0[e]; bfr[8 + e] = p1[e]; }
                acc = __builtin_amdgcn_wmma_f32_16x16x32_bf16(false, af[s2], false, bfr, (short)0, acc, false, false);
            }
            #pragma unroll
            for (int r = 0; r < 8; ++r)
                S[mr * 16 + r + 8 * kh][1 + jt + lm] = acc[r];
        }
    }
    __syncthreads();

    // ---- gcol + mask + softmax: one row per thread ----
    if (tid < 64) {
        const int qrow = tid;
        float g = 0.f;
        for (int d = 0; d < 64; ++d) g += (float)qs[qrow][d] * k0s[d];
        S[qrow][0] = g;
        float mx = g;
        for (int j = 0; j < 192; ++j) {
            int kglob = c * WBAND + j - WBAND;
            int rel = j - WBAND - qrow; if (rel < 0) rel = -rel;
            bool keep = (rel <= WBAND) && (kglob >= 1) && (kglob < NTOK0);
            float s = keep ? S[qrow][1 + j] : -1e9f;
            S[qrow][1 + j] = s;
            mx = fmaxf(mx, s);
        }
        float sum = 0.f;
        for (int j = 0; j < 193; ++j) {
            float e = __expf(S[qrow][j] - mx);
            S[qrow][j] = e;
            sum += e;
        }
        float inv = 1.f / sum;
        for (int j = 0; j < 193; ++j) S[qrow][j] *= inv;
    }
    __syncthreads();

    // ---- out = P[:,1:] @ v3 + P[:,0]*v0 : 64x64 = 4x4 tiles, 2 per wave ----
    {
        const int mr = wid & 3;
        const int cg = wid >> 2;
        for (int tt = 0; tt < 2; ++tt) {
            int nt = (cg * 2 + tt) * 16;
            v8f acc = {};
            for (int jb = 0; jb < 192; jb += 32) {
                v16bf af, bfr;
                const float* p0 = &S[mr * 16 + lm][1 + jb + kh * 8];
                const float* p1 = &S[mr * 16 + lm][1 + jb + 16 + kh * 8];
                #pragma unroll
                for (int e = 0; e < 8; ++e) { af[e] = (__bf16)p0[e]; af[8 + e] = (__bf16)p1[e]; }
                #pragma unroll
                for (int e = 0; e < 8; ++e) {
                    bfr[e]     = vs[jb + kh * 8 + e][nt + lm];
                    bfr[8 + e] = vs[jb + 16 + kh * 8 + e][nt + lm];
                }
                acc = __builtin_amdgcn_wmma_f32_16x16x32_bf16(false, af, false, bfr, (short)0, acc, false, false);
            }
            #pragma unroll
            for (int r = 0; r < 8; ++r) {
                int m = mr * 16 + r + 8 * kh;
                int n = nt + lm;
                out[base + (size_t)(c * WBAND + m) * DIM + n] = acc[r] + S[m][0] * v0s[n];
            }
        }
    }
}

// ---------------------------------------------------------------------------
// Global attention for token 0: one block per (h, b); overwrites out row 0.
// ---------------------------------------------------------------------------
__global__ __launch_bounds__(256)
void global_attn_kernel(const float* __restrict__ qg, const float* __restrict__ kg,
                        const float* __restrict__ vg, float* __restrict__ out)
{
    const int h = blockIdx.x, b = blockIdx.y;
    __shared__ float q0[64];
    __shared__ float sc[NTOK];
    __shared__ float red[256];
    const int tid = threadIdx.x;
    if (tid < 64) q0[tid] = qg[b * DIM + h * HD + tid] * 0.125f;
    __syncthreads();
    const size_t base = (size_t)b * NTOK * DIM + (size_t)h * HD;
    float lmax = -1e30f;
    for (int n = tid; n < NTOK; n += 256) {
        float s = -1e9f;
        if (n < NTOK0) {
            s = 0.f;
            const float* kr = kg + base + (size_t)n * DIM;
            for (int d = 0; d < 64; ++d) s += q0[d] * kr[d];
        }
        sc[n] = s;
        lmax = fmaxf(lmax, s);
    }
    red[tid] = lmax; __syncthreads();
    for (int o = 128; o > 0; o >>= 1) { if (tid < o) red[tid] = fmaxf(red[tid], red[tid + o]); __syncthreads(); }
    float mx = red[0]; __syncthreads();
    float lsum = 0.f;
    for (int n = tid; n < NTOK; n += 256) {
        float e = __expf(sc[n] - mx);
        sc[n] = e;
        lsum += e;
    }
    red[tid] = lsum; __syncthreads();
    for (int o = 128; o > 0; o >>= 1) { if (tid < o) red[tid] += red[tid + o]; __syncthreads(); }
    float inv = 1.f / red[0];
    __syncthreads();
    if (tid < 64) {
        float acc = 0.f;
        for (int n = 0; n < NTOK0; ++n) acc += sc[n] * vg[base + (size_t)n * DIM + tid];
        out[base + tid] = acc * inv;
    }
}

// ---------------------------------------------------------------------------
// Head: per-batch LN(cls) then [512]x[512,1000] + bias (tiny)
// ---------------------------------------------------------------------------
__global__ __launch_bounds__(256)
void head_kernel(const float* __restrict__ x, const float* __restrict__ lw,
                 const float* __restrict__ lb, const float* __restrict__ hw,
                 const float* __restrict__ hb, float* __restrict__ outp)
{
    const int b = blockIdx.x;
    __shared__ float xr[DIM];
    __shared__ float red[256];
    const int tid = threadIdx.x;
    const float* row = x + (size_t)b * NTOK * DIM;
    float v0 = row[tid], v1 = row[tid + 256];
    red[tid] = v0 + v1; __syncthreads();
    for (int o = 128; o > 0; o >>= 1) { if (tid < o) red[tid] += red[tid + o]; __syncthreads(); }
    float mean = red[0] * (1.f / DIM); __syncthreads();
    float d0 = v0 - mean, d1 = v1 - mean;
    red[tid] = d0 * d0 + d1 * d1; __syncthreads();
    for (int o = 128; o > 0; o >>= 1) { if (tid < o) red[tid] += red[tid + o]; __syncthreads(); }
    float rs = rsqrtf(red[0] * (1.f / DIM) + 1e-5f);
    xr[tid]       = d0 * rs * lw[tid]       + lb[tid];
    xr[tid + 256] = d1 * rs * lw[tid + 256] + lb[tid + 256];
    __syncthreads();
    for (int j = tid; j < NCLS; j += 256) {
        float acc = hb[j];
        for (int d = 0; d < DIM; ++d) acc += xr[d] * hw[(size_t)d * NCLS + j];
        outp[b * NCLS + j] = acc;
    }
}

// ---------------------------------------------------------------------------
extern "C" void kernel_launch(void* const* d_in, const int* in_sizes, int n_in,
                              void* d_out, int out_size, void* d_ws, size_t ws_size,
                              hipStream_t stream)
{
    (void)in_sizes; (void)n_in; (void)out_size; (void)ws_size;
    const float* img      = (const float*)d_in[0];
    const float* patch_w  = (const float*)d_in[1];
    const float* patch_b  = (const float*)d_in[2];
    const float* cls_tok  = (const float*)d_in[3];
    const float* pos_emb  = (const float*)d_in[4];
    const float* ln1_w    = (const float*)d_in[5];
    const float* ln1_b    = (const float*)d_in[6];
    const float* wq       = (const float*)d_in[7];
    const float* bq       = (const float*)d_in[8];
    const float* wk       = (const float*)d_in[9];
    const float* bk       = (const float*)d_in[10];
    const float* wv       = (const float*)d_in[11];
    const float* bv       = (const float*)d_in[12];
    const float* wqg      = (const float*)d_in[13];
    const float* bqg      = (const float*)d_in[14];
    const float* wkg      = (const float*)d_in[15];
    const float* bkg      = (const float*)d_in[16];
    const float* wvg      = (const float*)d_in[17];
    const float* bvg      = (const float*)d_in[18];
    const float* ln2_w    = (const float*)d_in[19];
    const float* ln2_b    = (const float*)d_in[20];
    const float* w1       = (const float*)d_in[21];
    const float* b1       = (const float*)d_in[22];
    const float* w2       = (const float*)d_in[23];
    const float* b2       = (const float*)d_in[24];
    const float* hlw      = (const float*)d_in[25];
    const float* hlb      = (const float*)d_in[26];
    const float* head_w   = (const float*)d_in[27];
    const float* head_b   = (const float*)d_in[28];

    const size_t MN = (size_t)BBATCH * NTOK * DIM;   // 4.72M floats
    float* x    = (float*)d_ws;
    float* xln  = x + MN;
    float* attn = xln + MN;          // also temp for patch embeddings
    float* big  = attn + MN;         // 5*MN region, aliased by xcol / hbuf
    float* qb   = big;
    float* kb   = big + MN;
    float* vb   = big + 2 * MN;
    float* kgb  = big + 3 * MN;
    float* vgb  = big + 4 * MN;
    float* xcol = big;               // alias: used only before layers (25.2MB)
    float* hbuf = big;               // alias: used only during MLP (75.5MB)
    float* qgb  = big + 5 * MN;      // 8*512 floats

    // ---- patch embed ----
    {
        size_t tot = (size_t)BBATCH * NPATCH * KPATCH;
        im2col_kernel<<<(unsigned)((tot + 255) / 256), 256, 0, stream>>>(img, xcol);
        dim3 g(DIM / 64, (BBATCH * NPATCH + 127) / 128);
        gemm_bf16_wmma<0, 0><<<g, 256, 0, stream>>>(xcol, patch_w, patch_b, attn,
                                                    BBATCH * NPATCH, KPATCH, DIM);
        assemble_kernel<<<(unsigned)((MN + 255) / 256), 256, 0, stream>>>(attn, cls_tok, pos_emb, x);
    }

    const int Mtok = BBATCH * NTOK;  // 9216
    for (int L = 0; L < DEPTHL; ++L) {
        const size_t wo = (size_t)L * DIM * DIM;
        layernorm_kernel<<<Mtok, 256, 0, stream>>>(x, ln1_w + L * DIM, ln1_b + L * DIM, xln);
        dim3 g(DIM / 64, (Mtok + 127) / 128);
        gemm_bf16_wmma<0, 0><<<g, 256, 0, stream>>>(xln, wq  + wo, bq  + L * DIM, qb,  Mtok, DIM, DIM);
        gemm_bf16_wmma<0, 0><<<g, 256, 0, stream>>>(xln, wk  + wo, bk  + L * DIM, kb,  Mtok, DIM, DIM);
        gemm_bf16_wmma<0, 0><<<g, 256, 0, stream>>>(xln, wv  + wo, bv  + L * DIM, vb,  Mtok, DIM, DIM);
        gemm_bf16_wmma<0, 0><<<g, 256, 0, stream>>>(xln, wkg + wo, bkg + L * DIM, kgb, Mtok, DIM, DIM);
        gemm_bf16_wmma<0, 0><<<g, 256, 0, stream>>>(xln, wvg + wo, bvg + L * DIM, vgb, Mtok, DIM, DIM);
        qg_proj_kernel<<<BBATCH, 256, 0, stream>>>(xln, wqg + wo, bqg + L * DIM, qgb);

        dim3 gb(NCH, HEADS, BBATCH);
        band_attn_kernel<<<gb, 256, 0, stream>>>(qb, kb, vb, attn);
        dim3 gg(HEADS, BBATCH);
        global_attn_kernel<<<gg, 256, 0, stream>>>(qgb, kgb, vgb, attn);
        add_kernel<<<(unsigned)((MN + 255) / 256), 256, 0, stream>>>(x, attn, MN);

        layernorm_kernel<<<Mtok, 256, 0, stream>>>(x, ln2_w + L * DIM, ln2_b + L * DIM, xln);
        dim3 g1(MLPD / 64, (Mtok + 127) / 128);
        gemm_bf16_wmma<1, 0><<<g1, 256, 0, stream>>>(xln, w1 + (size_t)L * DIM * MLPD,
                                                     b1 + (size_t)L * MLPD, hbuf, Mtok, DIM, MLPD);
        dim3 g2(DIM / 64, (Mtok + 127) / 128);
        gemm_bf16_wmma<0, 1><<<g2, 256, 0, stream>>>(hbuf, w2 + (size_t)L * MLPD * DIM,
                                                     b2 + (size_t)L * DIM, x, Mtok, MLPD, DIM);
    }

    head_kernel<<<BBATCH, 256, 0, stream>>>(x, hlw, hlb, head_w, head_b, (float*)d_out);
}
